// MiniFluxDiT_10642928959657
// MI455X (gfx1250) — matrix operand order repaired and checked by reading, not compiled, not run on hardware
//
#include <hip/hip_runtime.h>
#include <hip/hip_bf16.h>
#include <math.h>

// ---------------------------------------------------------------------------
// Types for CDNA5 WMMA
// ---------------------------------------------------------------------------
typedef __bf16 bf16_t;
typedef __attribute__((ext_vector_type(16))) __bf16 v16bf;
typedef __attribute__((ext_vector_type(8)))  float  v8f;
typedef __attribute__((ext_vector_type(4)))  int    v4i_t;

#define CDIV(a, b) (((a) + (b) - 1) / (b))

// Model dims (MiniFlux DiT reference)
#define BB   2
#define CC   4
#define HH   64
#define WW   64
#define PP   2
#define EE   1024
#define SS_  1024      // image tokens
#define LL   2
#define NHh  16
#define HD   64
#define SKk  256       // qwen context length
#define KDd  1536
#define II   4096
#define NEe  8
#define CLIPD 768
#define TED  256
#define ADA  (9 * EE)

// ---------------------------------------------------------------------------
// Device helpers
// ---------------------------------------------------------------------------
__device__ __forceinline__ float gelu_tanh(float x) {
    return 0.5f * x * (1.0f + tanhf(0.7978845608028654f * (x + 0.044715f * x * x * x)));
}
__device__ __forceinline__ float silu(float x) {
    return x / (1.0f + expf(-x));
}

// ---- CDNA5 async global->LDS staging (ASYNCcnt path), with safe fallback ----
typedef __attribute__((address_space(1))) v4i_t* as1_v4i;
typedef __attribute__((address_space(3))) v4i_t* as3_v4i;

#if __has_builtin(__builtin_amdgcn_global_load_async_to_lds_b128)
#define ASYNC_LDS 1
__device__ __forceinline__ void async_ld_b128(const float* g, float* l) {
    __builtin_amdgcn_global_load_async_to_lds_b128(
        (as1_v4i)(float*)g, (as3_v4i)l, 0, 0);
}
#else
#define ASYNC_LDS 0
__device__ __forceinline__ void async_ld_b128(const float* g, float* l) {
    *(float4*)l = *(const float4*)g;
}
#endif

__device__ __forceinline__ void wait_async_lds() {
#if ASYNC_LDS
#if __has_builtin(__builtin_amdgcn_s_wait_asynccnt)
    __builtin_amdgcn_s_wait_asynccnt(0);
#else
    asm volatile("s_wait_asynccnt 0" ::: "memory");
#endif
#endif
}

// ---------------------------------------------------------------------------
// WMMA GEMM:  C[m,n] (+)= rowscale[m] * act( alpha * sum_k A[m,k]*B[n,k] + bias[n] )
// A: f32 [M x K] row-major (K contiguous), row stride lda
// B: f32 [N x K] row-major (K contiguous), row stride ldb   (i.e. C = A * B^T)
// Converted to bf16 in-register; f32 accumulate via v_wmma_f32_16x16x32_bf16.
// Each wave computes a 64x64 macro-tile: 4x4 subtiles -> 16 WMMAs / K-step.
// Batched over blockIdx.z; z decomposed as zo = z / zInner, zi = z % zInner,
// per-operand offsets  zo*Outer + zi*Inner  (covers [b,h] attention batching).
// Requires M % 64 == 0, N % 64 == 0, K % 32 == 0 (true for every call site).
// ---------------------------------------------------------------------------
__global__ __launch_bounds__(256) void gemm_wmma_bf16(
    const float* __restrict__ A, long lda, long aOuter, long aInner,
    const float* __restrict__ Bm, long ldb, long bOuter, long bInner,
    const float* __restrict__ bias,
    float* __restrict__ C, long ldc, long cOuter, long cInner,
    int M, int N, int K, int zInner,
    float alpha, int act,
    const float* __restrict__ rowscale, int rsStride,
    int accumulate)
{
    const int z  = blockIdx.z;
    const int zo = z / zInner;
    const int zi = z - zo * zInner;
    A  += (long)zo * aOuter + (long)zi * aInner;
    Bm += (long)zo * bOuter + (long)zi * bInner;
    C  += (long)zo * cOuter + (long)zi * cInner;

    const int lane  = threadIdx.x & 31;
    const int wave  = threadIdx.x >> 5;
    const int lhalf = lane >> 4;     // 0 or 1
    const int lmod  = lane & 15;

    const int tilesM = M >> 6;       // 64-row macro tiles
    const int tilesN = N >> 6;       // 64-col macro tiles
    const int tileId = blockIdx.x * 8 + wave;
    if (tileId >= tilesM * tilesN) return;
    const int tm   = tileId % tilesM;
    const int tn   = tileId / tilesM;
    const int row0 = tm << 6;
    const int col0 = tn << 6;

    v8f acc[4][4];
#pragma unroll
    for (int mi = 0; mi < 4; mi++)
#pragma unroll
        for (int ni = 0; ni < 4; ni++) acc[mi][ni] = (v8f){0,0,0,0,0,0,0,0};

    // Per-lane base pointers matching CDNA5 16-bit A/B fragment layouts:
    //  A 16x32: lane<16 -> row M=lane, K in [0,8)u[16,24); lane>=16 -> K +8
    //  B 32x16: lane%16 = col N, K in [0,8)u[16,24); lane>=16 -> K +8
    const float* aRow[4];
#pragma unroll
    for (int mi = 0; mi < 4; mi++)
        aRow[mi] = A + (long)(row0 + mi * 16 + lmod) * lda + (lhalf << 3);
    const float* bRow[4];
#pragma unroll
    for (int ni = 0; ni < 4; ni++)
        bRow[ni] = Bm + (long)(col0 + ni * 16 + lmod) * ldb + (lhalf << 3);

    for (int kk = 0; kk < K; kk += 32) {
        // prefetch next K-slab (speculative; dropped past end)
        __builtin_prefetch(aRow[0] + kk + 64);
        __builtin_prefetch(bRow[0] + kk + 64);

        v16bf af[4];
#pragma unroll
        for (int mi = 0; mi < 4; mi++) {
            const float* p = aRow[mi] + kk;
            float4 c0 = *(const float4*)(p);
            float4 c1 = *(const float4*)(p + 4);
            float4 c2 = *(const float4*)(p + 16);
            float4 c3 = *(const float4*)(p + 20);
            v16bf f;
            f[0]=(bf16_t)c0.x; f[1]=(bf16_t)c0.y; f[2]=(bf16_t)c0.z; f[3]=(bf16_t)c0.w;
            f[4]=(bf16_t)c1.x; f[5]=(bf16_t)c1.y; f[6]=(bf16_t)c1.z; f[7]=(bf16_t)c1.w;
            f[8]=(bf16_t)c2.x; f[9]=(bf16_t)c2.y; f[10]=(bf16_t)c2.z; f[11]=(bf16_t)c2.w;
            f[12]=(bf16_t)c3.x; f[13]=(bf16_t)c3.y; f[14]=(bf16_t)c3.z; f[15]=(bf16_t)c3.w;
            af[mi] = f;
        }
        v16bf bfr[4];
#pragma unroll
        for (int ni = 0; ni < 4; ni++) {
            const float* p = bRow[ni] + kk;
            float4 c0 = *(const float4*)(p);
            float4 c1 = *(const float4*)(p + 4);
            float4 c2 = *(const float4*)(p + 16);
            float4 c3 = *(const float4*)(p + 20);
            v16bf f;
            f[0]=(bf16_t)c0.x; f[1]=(bf16_t)c0.y; f[2]=(bf16_t)c0.z; f[3]=(bf16_t)c0.w;
            f[4]=(bf16_t)c1.x; f[5]=(bf16_t)c1.y; f[6]=(bf16_t)c1.z; f[7]=(bf16_t)c1.w;
            f[8]=(bf16_t)c2.x; f[9]=(bf16_t)c2.y; f[10]=(bf16_t)c2.z; f[11]=(bf16_t)c2.w;
            f[12]=(bf16_t)c3.x; f[13]=(bf16_t)c3.y; f[14]=(bf16_t)c3.z; f[15]=(bf16_t)c3.w;
            bfr[ni] = f;
        }
#pragma unroll
        for (int mi = 0; mi < 4; mi++)
#pragma unroll
            for (int ni = 0; ni < 4; ni++)
                acc[mi][ni] = __builtin_amdgcn_wmma_f32_16x16x32_bf16(
                    false, af[mi], false, bfr[ni], (short)0, acc[mi][ni], false, false);
    }

    // Epilogue: C 16x16 layout -> VGPR r, lane: row = 8*lhalf + r, col = lmod
#pragma unroll
    for (int mi = 0; mi < 4; mi++) {
#pragma unroll
        for (int ni = 0; ni < 4; ni++) {
            const int col = col0 + ni * 16 + lmod;
            const float bsv = bias ? bias[col] : 0.0f;
#pragma unroll
            for (int r = 0; r < 8; r++) {
                const int row = row0 + mi * 16 + lhalf * 8 + r;
                float v = acc[mi][ni][r] * alpha + bsv;
                if (act == 1) v = gelu_tanh(v);
                if (rowscale) v *= rowscale[(long)row * rsStride];
                float* cp = C + (long)row * ldc + col;
                if (accumulate) *cp += v; else *cp = v;
            }
        }
    }
}

// ---------------------------------------------------------------------------
// LayerNorm (+ optional adaLN modulate):  y = ln(x)*(1+sc[b]) + sh[b]
// One block (256 thr) per row of length E<=1024. Row staged into LDS via
// async global->LDS copy; LDS tree reductions.
// ---------------------------------------------------------------------------
__global__ __launch_bounds__(256) void ln_mod_kernel(
    const float* __restrict__ x,
    const float* __restrict__ sh, const float* __restrict__ sc, int modStride,
    float* __restrict__ y, int E, int S)
{
    const int row = blockIdx.x;      // b*S + s
    const int b   = row / S;
    const int tid = threadIdx.x;
    const float* xr = x + (long)row * E;
    float* yr = y + (long)row * E;

    __shared__ float rowb[1024];
    __shared__ float red0[256];
    __shared__ float red1[256];

    for (int j = tid * 4; j < E; j += 1024) async_ld_b128(xr + j, rowb + j);
    wait_async_lds();
    __syncthreads();

    float s0 = 0.f, s1 = 0.f;
    for (int j = tid; j < E; j += 256) { float v = rowb[j]; s0 += v; s1 += v * v; }
    red0[tid] = s0; red1[tid] = s1;
    __syncthreads();
    for (int o = 128; o > 0; o >>= 1) {
        if (tid < o) { red0[tid] += red0[tid + o]; red1[tid] += red1[tid + o]; }
        __syncthreads();
    }
    const float mean = red0[0] / (float)E;
    const float var  = red1[0] / (float)E - mean * mean;
    const float rstd = rsqrtf(var + 1e-6f);

    for (int j = tid; j < E; j += 256) {
        float n = (rowb[j] - mean) * rstd;
        if (sh) {
            const float shv = sh[(long)b * modStride + j];
            const float scv = sc[(long)b * modStride + j];
            n = n * (1.0f + scv) + shv;
        }
        yr[j] = n;
    }
}

// ---------------------------------------------------------------------------
// Row softmax in place, row staged via async global->LDS (single global read).
// One block per row, Lrow <= 1024.
// ---------------------------------------------------------------------------
__global__ __launch_bounds__(256) void softmax_kernel(float* __restrict__ s, int Lrow)
{
    float* p = s + (long)blockIdx.x * Lrow;
    const int tid = threadIdx.x;
    __shared__ float rowb[1024];
    __shared__ float red[256];

    for (int j = tid * 4; j < Lrow; j += 1024) async_ld_b128(p + j, rowb + j);
    wait_async_lds();
    __syncthreads();

    float m = -1e30f;
    for (int j = tid; j < Lrow; j += 256) m = fmaxf(m, rowb[j]);
    red[tid] = m; __syncthreads();
    for (int o = 128; o > 0; o >>= 1) {
        if (tid < o) red[tid] = fmaxf(red[tid], red[tid + o]);
        __syncthreads();
    }
    m = red[0]; __syncthreads();

    float sum = 0.f;
    for (int j = tid; j < Lrow; j += 256) { float e = expf(rowb[j] - m); rowb[j] = e; sum += e; }
    red[tid] = sum; __syncthreads();
    for (int o = 128; o > 0; o >>= 1) {
        if (tid < o) red[tid] += red[tid + o];
        __syncthreads();
    }
    const float inv = 1.0f / red[0];
    __syncthreads();
    for (int j = tid; j < Lrow; j += 256) p[j] = rowb[j] * inv;
}

// vt[((b*NH+h)*HD+d)*T + t] = v[(b*T+t)*ldv + vOff + h*HD + d]
__global__ __launch_bounds__(256) void transpose_v_kernel(
    const float* __restrict__ v, float* __restrict__ vt, int T, long ldv, int vOff)
{
    long idx = (long)blockIdx.x * 256 + threadIdx.x;
    const long total = (long)BB * NHh * HD * T;
    if (idx >= total) return;
    int t = (int)(idx % T);
    int d = (int)((idx / T) % HD);
    int h = (int)((idx / ((long)T * HD)) % NHh);
    int b = (int)(idx / ((long)T * HD * NHh));
    vt[idx] = v[(long)(b * T + t) * ldv + vOff + h * HD + d];
}

// xt += g[b,:] * y  (g row stride = gStride)
__global__ __launch_bounds__(256) void residual_gate_kernel(
    float* __restrict__ xt, const float* __restrict__ y,
    const float* __restrict__ g, int gStride)
{
    long idx = (long)blockIdx.x * 256 + threadIdx.x;
    const long total = (long)BB * SS_ * EE;
    if (idx >= total) return;
    const int e = (int)(idx % EE);
    const int b = (int)(idx / ((long)SS_ * EE));
    xt[idx] += g[(long)b * gStride + e] * y[idx];
}

__global__ __launch_bounds__(256) void zero_kernel(float* __restrict__ p, long n)
{
    long idx = (long)blockIdx.x * 256 + threadIdx.x;
    if (idx < n) p[idx] = 0.0f;
}

// patch-embed: xt[b,s,e] = patches[b,s,:] . conv_w[e,:] + conv_b[e] + pos[s,e]
__global__ __launch_bounds__(256) void patch_embed_kernel(
    const float* __restrict__ x, const float* __restrict__ cw,
    const float* __restrict__ cb, const float* __restrict__ pos,
    float* __restrict__ xt)
{
    long idx = (long)blockIdx.x * 256 + threadIdx.x;
    const long total = (long)BB * SS_ * EE;
    if (idx >= total) return;
    const int e = (int)(idx % EE);
    const int s = (int)((idx / EE) % SS_);
    const int b = (int)(idx / ((long)EE * SS_));
    const int hy = s >> 5, wx = s & 31;
    float acc = cb[e];
    const float* wrow = cw + (long)e * (CC * PP * PP);
#pragma unroll
    for (int c = 0; c < CC; c++)
#pragma unroll
        for (int pi = 0; pi < PP; pi++)
#pragma unroll
            for (int pj = 0; pj < PP; pj++) {
                const float xv = x[(((long)(b * CC + c) * HH) + hy * PP + pi) * WW + wx * PP + pj];
                acc += xv * wrow[c * PP * PP + pi * PP + pj];
            }
    xt[idx] = acc + pos[(long)s * EE + e];
}

// sinusoidal timestep embedding + 2-layer MLP (dim 256). One block per batch.
__global__ __launch_bounds__(256) void time_embed_kernel(
    const float* __restrict__ t,
    const float* __restrict__ w1, const float* __restrict__ b1,
    const float* __restrict__ w2, const float* __restrict__ b2,
    float* __restrict__ te)
{
    __shared__ float e0[TED];
    __shared__ float h1[TED];
    const int b = blockIdx.x, tid = threadIdx.x;
    const float tv = t[b];
    if (tid < 128) {
        const float f = expf(-9.210340371976184f / 127.0f * (float)tid);
        const float a = tv * f;
        e0[tid]       = sinf(a);
        e0[tid + 128] = cosf(a);
    }
    __syncthreads();
    {
        float s = b1[tid];
        const float* wr = w1 + (long)tid * TED;
        for (int k = 0; k < TED; k++) s += e0[k] * wr[k];
        h1[tid] = silu(s);
    }
    __syncthreads();
    {
        float s = b2[tid];
        const float* wr = w2 + (long)tid * TED;
        for (int k = 0; k < TED; k++) s += h1[k] * wr[k];
        te[(long)b * TED + tid] = s;
    }
}

// cs[b,e] = silu( concat(te[b], clip[b]) . cond_w[e,:] + cond_b[e] )
__global__ __launch_bounds__(256) void cond_kernel(
    const float* __restrict__ te, const float* __restrict__ clip,
    const float* __restrict__ cw, const float* __restrict__ cb,
    float* __restrict__ cs)
{
    long idx = (long)blockIdx.x * 256 + threadIdx.x;
    if (idx >= (long)BB * EE) return;
    const int e = (int)(idx % EE);
    const int b = (int)(idx / EE);
    const float* wr = cw + (long)e * (TED + CLIPD);
    float acc = cb[e];
    for (int k = 0; k < TED; k++)   acc += te[(long)b * TED + k] * wr[k];
    for (int k = 0; k < CLIPD; k++) acc += clip[(long)b * CLIPD + k] * wr[TED + k];
    cs[idx] = silu(acc);
}

// pr[b,j] = cs[b,:] . ada_w[j,:] + ada_b[j]
__global__ __launch_bounds__(256) void ada_proj_kernel(
    const float* __restrict__ cs, const float* __restrict__ aw,
    const float* __restrict__ ab, float* __restrict__ pr)
{
    long idx = (long)blockIdx.x * 256 + threadIdx.x;
    if (idx >= (long)BB * ADA) return;
    const int j = (int)(idx % ADA);
    const int b = (int)(idx / ADA);
    const float* wr = aw + (long)j * EE;
    const float* cr = cs + (long)b * EE;
    float acc = ab[j];
    for (int k = 0; k < EE; k++) acc += cr[k] * wr[k];
    pr[idx] = acc;
}

// gate logits: gl[token,e] = xn[token,:] . gw[e,:] + gb[e]
__global__ __launch_bounds__(256) void gate_logits_kernel(
    const float* __restrict__ xn, const float* __restrict__ gw,
    const float* __restrict__ gb, float* __restrict__ gl)
{
    long idx = (long)blockIdx.x * 256 + threadIdx.x;
    if (idx >= (long)BB * SS_ * NEe) return;
    const int e = (int)(idx % NEe);
    const long tkn = idx / NEe;
    const float* xr = xn + tkn * EE;
    const float* wr = gw + (long)e * EE;
    float acc = gb[e];
    for (int k = 0; k < EE; k++) acc += xr[k] * wr[k];
    gl[idx] = acc;
}

// per-token softmax over 8 experts, top-2, renormalize -> dense wd[token,8]
__global__ __launch_bounds__(256) void route_kernel(
    const float* __restrict__ gl, float* __restrict__ wd)
{
    long tkn = (long)blockIdx.x * 256 + threadIdx.x;
    if (tkn >= (long)BB * SS_) return;
    const float* p = gl + tkn * NEe;
    float m = -1e30f;
#pragma unroll
    for (int e = 0; e < NEe; e++) m = fmaxf(m, p[e]);
    float pe[NEe]; float sum = 0.f;
#pragma unroll
    for (int e = 0; e < NEe; e++) { pe[e] = expf(p[e] - m); sum += pe[e]; }
    const float inv = 1.0f / sum;
#pragma unroll
    for (int e = 0; e < NEe; e++) pe[e] *= inv;
    int i1 = 0;
#pragma unroll
    for (int e = 1; e < NEe; e++) if (pe[e] > pe[i1]) i1 = e;
    int i2 = (i1 == 0) ? 1 : 0;
#pragma unroll
    for (int e = 0; e < NEe; e++) if (e != i1 && pe[e] > pe[i2]) i2 = e;
    const float t1 = pe[i1], t2 = pe[i2];
    const float r = 1.0f / (t1 + t2 + 1e-6f);
    float* o = wd + tkn * NEe;
#pragma unroll
    for (int e = 0; e < NEe; e++) o[e] = 0.0f;
    o[i1] = t1 * r; o[i2] = t2 * r;
}

// final linear: fin[b,s,o] = ln(xt)[b,s,:] . fw[o,:] + fb[o]   (o < 16)
__global__ __launch_bounds__(256) void final_lin_kernel(
    const float* __restrict__ xn, const float* __restrict__ fw,
    const float* __restrict__ fb, float* __restrict__ fin)
{
    long idx = (long)blockIdx.x * 256 + threadIdx.x;
    const int OD = PP * PP * CC; // 16
    if (idx >= (long)BB * SS_ * OD) return;
    const int o = (int)(idx % OD);
    const long row = idx / OD;
    const float* xr = xn + row * EE;
    const float* wr = fw + (long)o * EE;
    float acc = fb[o];
    for (int k = 0; k < EE; k++) acc += xr[k] * wr[k];
    fin[idx] = acc;
}

// unpatchify: out[b,c,hy*2+pi,wx*2+pj] = fin[b, hy*32+wx, (pi*2+pj)*C + c]
__global__ __launch_bounds__(256) void unpatchify_kernel(
    const float* __restrict__ fin, float* __restrict__ out)
{
    long idx = (long)blockIdx.x * 256 + threadIdx.x;
    const long total = (long)BB * CC * HH * WW;
    if (idx >= total) return;
    const int wp = (int)(idx % WW);
    const int hp = (int)((idx / WW) % HH);
    const int c  = (int)((idx / ((long)WW * HH)) % CC);
    const int b  = (int)(idx / ((long)WW * HH * CC));
    const int hy = hp >> 1, pi = hp & 1;
    const int wx = wp >> 1, pj = wp & 1;
    const int s  = hy * 32 + wx;
    out[idx] = fin[((long)(b * SS_ + s)) * 16 + (pi * 2 + pj) * CC + c];
}

// ---------------------------------------------------------------------------
// Host side
// ---------------------------------------------------------------------------
static inline void launch_gemm(
    hipStream_t st,
    const float* A, long lda, long aO, long aI,
    const float* B, long ldb, long bO, long bI,
    const float* bias,
    float* C, long ldc, long cO, long cI,
    int M, int N, int K, int Z, int zInner,
    float alpha, int act,
    const float* rowscale, int rsStride, int accumulate)
{
    const int tiles = (M / 64) * (N / 64);
    dim3 grid(CDIV(tiles, 8), 1, Z);
    gemm_wmma_bf16<<<grid, 256, 0, st>>>(
        A, lda, aO, aI, B, ldb, bO, bI, bias,
        C, ldc, cO, cI, M, N, K, zInner, alpha, act,
        rowscale, rsStride, accumulate);
}

extern "C" void kernel_launch(void* const* d_in, const int* in_sizes, int n_in,
                              void* d_out, int out_size, void* d_ws, size_t ws_size,
                              hipStream_t stream)
{
    const float* x        = (const float*)d_in[0];
    const float* t        = (const float*)d_in[1];
    const float* clip_vec = (const float*)d_in[2];
    const float* qwen     = (const float*)d_in[3];
    const float* conv_w   = (const float*)d_in[4];
    const float* conv_b   = (const float*)d_in[5];
    const float* pos      = (const float*)d_in[6];
    const float* tw1      = (const float*)d_in[7];
    const float* tb1      = (const float*)d_in[8];
    const float* tw2      = (const float*)d_in[9];
    const float* tb2      = (const float*)d_in[10];
    const float* cond_w   = (const float*)d_in[11];
    const float* cond_b   = (const float*)d_in[12];
    const float* ada_w    = (const float*)d_in[13];
    const float* ada_b    = (const float*)d_in[14];
    const float* sa_in_w  = (const float*)d_in[15];
    const float* sa_in_b  = (const float*)d_in[16];
    const float* sa_out_w = (const float*)d_in[17];
    const float* sa_out_b = (const float*)d_in[18];
    const float* ca_q_w   = (const float*)d_in[19];
    const float* ca_k_w   = (const float*)d_in[20];
    const float* ca_v_w   = (const float*)d_in[21];
    const float* ca_in_b  = (const float*)d_in[22];
    const float* ca_out_w = (const float*)d_in[23];
    const float* ca_out_b = (const float*)d_in[24];
    const float* gate_w   = (const float*)d_in[25];
    const float* gate_b   = (const float*)d_in[26];
    const float* e_w1     = (const float*)d_in[27];
    const float* e_b1     = (const float*)d_in[28];
    const float* e_w2     = (const float*)d_in[29];
    const float* e_b2     = (const float*)d_in[30];
    const float* fin_w    = (const float*)d_in[31];
    const float* fin_b    = (const float*)d_in[32];

    // ---- workspace bump allocator (f32 buffers) ----
    char*  base = (char*)d_ws;
    size_t off  = 0;
    auto alloc = [&](long nfloat) -> float* {
        float* p = (float*)(base + off);
        off += ((size_t)nfloat * 4 + 255) & ~(size_t)255;
        return p;
    };
    float* xt    = alloc((long)BB * SS_ * EE);
    float* xn    = alloc((long)BB * SS_ * EE);
    float* qkv   = alloc((long)BB * SS_ * 3 * EE);
    float* vt    = alloc((long)BB * NHh * HD * SS_);
    float* sc_sa = alloc((long)BB * NHh * SS_ * SS_);
    float* obuf  = alloc((long)BB * SS_ * EE);
    float* ybuf  = alloc((long)BB * SS_ * EE);
    float* qb    = alloc((long)BB * SS_ * EE);
    float* kb    = alloc((long)BB * SKk * EE);
    float* vb    = alloc((long)BB * SKk * EE);
    float* vtc   = alloc((long)BB * NHh * HD * SKk);
    float* sc_ca = alloc((long)BB * NHh * SS_ * SKk);
    float* hid   = alloc((long)BB * SS_ * II);
    float* moe   = alloc((long)BB * SS_ * EE);
    float* pr    = alloc((long)BB * ADA);
    float* cs    = alloc((long)BB * EE);
    float* te    = alloc((long)BB * TED);
    float* gl    = alloc((long)BB * SS_ * NEe);
    float* wd    = alloc((long)BB * SS_ * NEe);
    float* fin   = alloc((long)BB * SS_ * 16);
    (void)ws_size; (void)in_sizes; (void)n_in; (void)out_size;

    const long tokE = (long)BB * SS_ * EE;
    const int  M    = BB * SS_;          // 2048 rows of image tokens
    const int  Mk   = BB * SKk;          // 512 rows of context tokens
    const int  Zbh  = BB * NHh;          // 32 (b,h) attention batches

    // ---- preamble ----
    patch_embed_kernel<<<CDIV(tokE, 256), 256, 0, stream>>>(x, conv_w, conv_b, pos, xt);
    time_embed_kernel<<<BB, 256, 0, stream>>>(t, tw1, tb1, tw2, tb2, te);
    cond_kernel<<<CDIV((long)BB * EE, 256), 256, 0, stream>>>(te, clip_vec, cond_w, cond_b, cs);

    for (int i = 0; i < LL; i++) {
        const float* adaW = ada_w + (long)i * ADA * EE;
        const float* adaB = ada_b + (long)i * ADA;
        ada_proj_kernel<<<CDIV((long)BB * ADA, 256), 256, 0, stream>>>(cs, adaW, adaB, pr);
        const float* sh_sa = pr + 0 * EE, *sc_samod = pr + 1 * EE, *g_sa = pr + 2 * EE;
        const float* sh_ca = pr + 3 * EE, *sc_camod = pr + 4 * EE, *g_ca = pr + 5 * EE;
        const float* sh_m  = pr + 6 * EE, *sc_mmod  = pr + 7 * EE, *g_m  = pr + 8 * EE;

        // ================= self-attention =================
        ln_mod_kernel<<<M, 256, 0, stream>>>(xt, sh_sa, sc_samod, ADA, xn, EE, SS_);
        launch_gemm(stream, xn, EE, 0, 0,
                    sa_in_w + (long)i * 3 * EE * EE, EE, 0, 0,
                    sa_in_b + (long)i * 3 * EE,
                    qkv, 3 * EE, 0, 0, M, 3 * EE, EE, 1, 1, 1.0f, 0, nullptr, 0, 0);
        transpose_v_kernel<<<CDIV(tokE, 256), 256, 0, stream>>>(qkv, vt, SS_, 3 * EE, 2 * EE);
        // scores[b,h] = q . k^T / 8
        launch_gemm(stream,
                    qkv, 3 * EE, (long)SS_ * 3 * EE, HD,                 // A = q
                    qkv + EE, 3 * EE, (long)SS_ * 3 * EE, HD,            // B = k
                    nullptr,
                    sc_sa, SS_, (long)NHh * SS_ * SS_, (long)SS_ * SS_,  // C
                    SS_, SS_, HD, Zbh, NHh, 0.125f, 0, nullptr, 0, 0);
        softmax_kernel<<<Zbh * SS_, 256, 0, stream>>>(sc_sa, SS_);
        // o[b,h] = probs . v^T   (vt is [b,h,d,t] so this is again A*B^T)
        launch_gemm(stream,
                    sc_sa, SS_, (long)NHh * SS_ * SS_, (long)SS_ * SS_,
                    vt, SS_, (long)NHh * HD * SS_, (long)HD * SS_,
                    nullptr,
                    obuf, EE, (long)SS_ * EE, HD,
                    SS_, HD, SS_, Zbh, NHh, 1.0f, 0, nullptr, 0, 0);
        launch_gemm(stream, obuf, EE, 0, 0,
                    sa_out_w + (long)i * EE * EE, EE, 0, 0,
                    sa_out_b + (long)i * EE,
                    ybuf, EE, 0, 0, M, EE, EE, 1, 1, 1.0f, 0, nullptr, 0, 0);
        residual_gate_kernel<<<CDIV(tokE, 256), 256, 0, stream>>>(xt, ybuf, g_sa, ADA);

        // ================= cross-attention =================
        ln_mod_kernel<<<M, 256, 0, stream>>>(xt, sh_ca, sc_camod, ADA, xn, EE, SS_);
        launch_gemm(stream, xn, EE, 0, 0,
                    ca_q_w + (long)i * EE * EE, EE, 0, 0,
                    ca_in_b + (long)i * 3 * EE,
                    qb, EE, 0, 0, M, EE, EE, 1, 1, 1.0f, 0, nullptr, 0, 0);
        launch_gemm(stream, qwen, KDd, 0, 0,
                    ca_k_w + (long)i * EE * KDd, KDd, 0, 0,
                    ca_in_b + (long)i * 3 * EE + EE,
                    kb, EE, 0, 0, Mk, EE, KDd, 1, 1, 1.0f, 0, nullptr, 0, 0);
        launch_gemm(stream, qwen, KDd, 0, 0,
                    ca_v_w + (long)i * EE * KDd, KDd, 0, 0,
                    ca_in_b + (long)i * 3 * EE + 2 * EE,
                    vb, EE, 0, 0, Mk, EE, KDd, 1, 1, 1.0f, 0, nullptr, 0, 0);
        transpose_v_kernel<<<CDIV((long)BB * NHh * HD * SKk, 256), 256, 0, stream>>>(
            vb, vtc, SKk, EE, 0);
        launch_gemm(stream,
                    qb, EE, (long)SS_ * EE, HD,
                    kb, EE, (long)SKk * EE, HD,
                    nullptr,
                    sc_ca, SKk, (long)NHh * SS_ * SKk, (long)SS_ * SKk,
                    SS_, SKk, HD, Zbh, NHh, 0.125f, 0, nullptr, 0, 0);
        softmax_kernel<<<Zbh * SS_, 256, 0, stream>>>(sc_ca, SKk);
        launch_gemm(stream,
                    sc_ca, SKk, (long)NHh * SS_ * SKk, (long)SS_ * SKk,
                    vtc, SKk, (long)NHh * HD * SKk, (long)HD * SKk,
                    nullptr,
                    obuf, EE, (long)SS_ * EE, HD,
                    SS_, HD, SKk, Zbh, NHh, 1.0f, 0, nullptr, 0, 0);
        launch_gemm(stream, obuf, EE, 0, 0,
                    ca_out_w + (long)i * EE * EE, EE, 0, 0,
                    ca_out_b + (long)i * EE,
                    ybuf, EE, 0, 0, M, EE, EE, 1, 1, 1.0f, 0, nullptr, 0, 0);
        residual_gate_kernel<<<CDIV(tokE, 256), 256, 0, stream>>>(xt, ybuf, g_ca, ADA);

        // ================= MoE (dense combine, top-2 routing weights) ========
        ln_mod_kernel<<<M, 256, 0, stream>>>(xt, sh_m, sc_mmod, ADA, xn, EE, SS_);
        gate_logits_kernel<<<CDIV((long)M * NEe, 256), 256, 0, stream>>>(
            xn, gate_w + (long)i * NEe * EE, gate_b + (long)i * NEe, gl);
        route_kernel<<<CDIV((long)M, 256), 256, 0, stream>>>(gl, wd);
        zero_kernel<<<CDIV(tokE, 256), 256, 0, stream>>>(moe, tokE);
        for (int e = 0; e < NEe; e++) {
            const float* w1 = e_w1 + ((long)i * NEe + e) * (long)II * EE;
            const float* b1 = e_b1 + ((long)i * NEe + e) * II;
            const float* w2 = e_w2 + ((long)i * NEe + e) * (long)EE * II;
            const float* b2 = e_b2 + ((long)i * NEe + e) * EE;
            // hid = gelu(xn . w1^T + b1)
            launch_gemm(stream, xn, EE, 0, 0, w1, EE, 0, 0, b1,
                        hid, II, 0, 0, M, II, EE, 1, 1, 1.0f, /*gelu*/1, nullptr, 0, 0);
            // moe += wd[:,e] * (hid . w2^T + b2)
            launch_gemm(stream, hid, II, 0, 0, w2, II, 0, 0, b2,
                        moe, EE, 0, 0, M, EE, II, 1, 1, 1.0f, 0,
                        wd + e, NEe, /*accumulate*/1);
        }
        residual_gate_kernel<<<CDIV(tokE, 256), 256, 0, stream>>>(xt, moe, g_m, ADA);
    }

    // ---- final layer + unpatchify ----
    ln_mod_kernel<<<M, 256, 0, stream>>>(xt, nullptr, nullptr, 0, xn, EE, SS_);
    final_lin_kernel<<<CDIV((long)M * 16, 256), 256, 0, stream>>>(xn, fin_w, fin_b, fin);
    unpatchify_kernel<<<CDIV((long)BB * CC * HH * WW, 256), 256, 0, stream>>>(
        fin, (float*)d_out);
}
